// MLA_26199300505739
// MI455X (gfx1250) — matrix operand order repaired and checked by reading, not compiled, not run on hardware
//
#include <hip/hip_runtime.h>

// ---------------------------------------------------------------------------
// MLA forward for MI455X (gfx1250): bf16 WMMA everywhere, fp32 accumulation.
// B=2, S=2048, D=2048, NH=16, DH=128, DCQ=768, DCKV=512, HALF=64
// ---------------------------------------------------------------------------

typedef __attribute__((ext_vector_type(16))) __bf16 v16bf;
typedef __attribute__((ext_vector_type(8)))  __bf16 v8bf;
typedef __attribute__((ext_vector_type(8)))  float  v8f;

#define NB_B   2
#define NB_S   2048
#define NB_D   2048
#define NB_NH  16
#define NB_DH  128
#define NB_DCQ 768
#define NB_DCKV 512
#define NB_HALF 64
#define NB_M   (NB_B * NB_S)          // 4096 token rows
#define NB_EPS 1.1920929e-07f

// Assemble a v16bf WMMA fragment from two contiguous 8-element (16B) runs.
__device__ __forceinline__ v16bf load_frag16(const __bf16* p0, const __bf16* p1) {
  v8bf lo = *(const v8bf*)p0;
  v8bf hi = *(const v8bf*)p1;
  v16bf r;
#pragma unroll
  for (int i = 0; i < 8; ++i) { r[i] = lo[i]; r[8 + i] = hi[i]; }
  return r;
}

// ---------------------------------------------------------------------------
// Weight prep: fp32 (K x N, row-major) -> bf16 transposed (N x K, row-major).
// Makes every WMMA B-fragment a contiguous per-lane load.
// ---------------------------------------------------------------------------
__global__ void k_transpose_cast(const float* __restrict__ src,
                                 __bf16* __restrict__ dst, int K, int N) {
  long long e = (long long)blockIdx.x * blockDim.x + threadIdx.x;
  if (e >= (long long)K * N) return;
  int k = (int)(e / N), n = (int)(e % N);
  dst[(size_t)n * K + k] = (__bf16)src[e];
}

// ---------------------------------------------------------------------------
// RMSNorm: one block per row; fp32 in, bf16 out.
// ---------------------------------------------------------------------------
__global__ __launch_bounds__(256) void k_rms_bf16(const float* __restrict__ src,
                                                  const float* __restrict__ w,
                                                  __bf16* __restrict__ dst, int W) {
  __shared__ float red[256];
  const int row = blockIdx.x;
  const float* p = src + (size_t)row * W;
  float s = 0.f;
  for (int i = threadIdx.x; i < W; i += 256) { float v = p[i]; s += v * v; }
  red[threadIdx.x] = s;
  __syncthreads();
  for (int off = 128; off > 0; off >>= 1) {
    if (threadIdx.x < off) red[threadIdx.x] += red[threadIdx.x + off];
    __syncthreads();
  }
  const float r = rsqrtf(red[0] / (float)W + NB_EPS);
  __bf16* q = dst + (size_t)row * W;
  for (int i = threadIdx.x; i < W; i += 256) q[i] = (__bf16)(p[i] * r * w[i]);
}

// ---------------------------------------------------------------------------
// Generic bf16 GEMM:  C(MxN, f32) = A(MxK, bf16) * Bt(NxK, bf16)^T [+ resid]
// Block: 256 threads = 8 waves stacked in M -> 256(M) x 64(N) block tile.
// Wave tile: 32(M) x 64(N) = 2 A-frags x 4 B-frags = 8 WMMA per k-step of 32.
// K-loop unrolled by 2 with explicit ping-pong fragment buffers: loads for
// the other buffer are in flight while WMMAs consume the current one, and no
// register copies are needed to rotate the pipeline. Requires Kdim % 64 == 0.
// ---------------------------------------------------------------------------
__global__ __launch_bounds__(256) void k_gemm_bf16(const __bf16* __restrict__ A,
                                                   const __bf16* __restrict__ Bt,
                                                   float* __restrict__ C,
                                                   const float* __restrict__ resid,
                                                   int Mdim, int Ndim, int Kdim) {
  const int lane = threadIdx.x & 31;
  const int wave = threadIdx.x >> 5;
  const int m0 = blockIdx.y * 256 + wave * 32;
  const int n0 = blockIdx.x * 64;
  const int ls = lane & 15;
  const bool hiHalf = lane >= 16;
  const int half8 = hiHalf ? 8 : 0;
  const int half16 = hiHalf ? 16 : 0;

  v8f acc[2][4] = {};
  const __bf16* aRow0 = A + (size_t)(m0 + ls) * Kdim + half8;       // M sub-tile 0
  const __bf16* aRow1 = A + (size_t)(m0 + 16 + ls) * Kdim + half8;  // M sub-tile 1
  const __bf16* bRow[4];
#pragma unroll
  for (int t = 0; t < 4; ++t)
    bRow[t] = Bt + (size_t)(n0 + t * 16 + ls) * Kdim + half8;

  // Ping-pong fragment buffers (no copies: even steps use 0, odd use 1).
  v16bf a0[2], a1[2], bb[2][4];

  // Prologue: buffer 0 <- k = 0.
  a0[0] = load_frag16(aRow0, aRow0 + 16);
  a1[0] = load_frag16(aRow1, aRow1 + 16);
#pragma unroll
  for (int t = 0; t < 4; ++t) bb[0][t] = load_frag16(bRow[t], bRow[t] + 16);

  for (int k0 = 0; k0 < Kdim; k0 += 64) {
    // Buffer 1 <- k0 + 32 (always in range: Kdim % 64 == 0).
    {
      const int kn = k0 + 32;
      a0[1] = load_frag16(aRow0 + kn, aRow0 + kn + 16);
      a1[1] = load_frag16(aRow1 + kn, aRow1 + kn + 16);
#pragma unroll
      for (int t = 0; t < 4; ++t)
        bb[1][t] = load_frag16(bRow[t] + kn, bRow[t] + kn + 16);
    }
    // Compute on buffer 0 (k0).
#pragma unroll
    for (int t = 0; t < 4; ++t) {
      acc[0][t] = __builtin_amdgcn_wmma_f32_16x16x32_bf16(
          false, a0[0], false, bb[0][t], (short)0, acc[0][t], false, false);
      acc[1][t] = __builtin_amdgcn_wmma_f32_16x16x32_bf16(
          false, a1[0], false, bb[0][t], (short)0, acc[1][t], false, false);
    }
    // Buffer 0 <- k0 + 64 (next iteration's even step).
    if (k0 + 64 < Kdim) {
      const int kn = k0 + 64;
      a0[0] = load_frag16(aRow0 + kn, aRow0 + kn + 16);
      a1[0] = load_frag16(aRow1 + kn, aRow1 + kn + 16);
#pragma unroll
      for (int t = 0; t < 4; ++t)
        bb[0][t] = load_frag16(bRow[t] + kn, bRow[t] + kn + 16);
    }
    // Compute on buffer 1 (k0 + 32).
#pragma unroll
    for (int t = 0; t < 4; ++t) {
      acc[0][t] = __builtin_amdgcn_wmma_f32_16x16x32_bf16(
          false, a0[1], false, bb[1][t], (short)0, acc[0][t], false, false);
      acc[1][t] = __builtin_amdgcn_wmma_f32_16x16x32_bf16(
          false, a1[1], false, bb[1][t], (short)0, acc[1][t], false, false);
    }
  }

#pragma unroll
  for (int mt = 0; mt < 2; ++mt) {
#pragma unroll
    for (int t = 0; t < 4; ++t) {
      const int col = n0 + t * 16 + ls;
#pragma unroll
      for (int r = 0; r < 8; ++r) {
        const int row = m0 + mt * 16 + r + half8;
        const size_t idx = (size_t)row * Ndim + col;
        float v = acc[mt][t][r];
        if (resid) v += resid[idx];
        C[idx] = v;
      }
    }
  }
}

// ---------------------------------------------------------------------------
// Build Q [B,NH,S,128], K [B,NH,S,128] (RoPE on the second half; K rotary part
// broadcast across heads) and V transposed [B,NH,DH,S] for contiguous PV frags.
// ---------------------------------------------------------------------------
__global__ __launch_bounds__(256) void k_build_qkv(
    const float* __restrict__ qc, const float* __restrict__ qr,
    const float* __restrict__ kc, const float* __restrict__ kr,
    const float* __restrict__ vraw, const float* __restrict__ cosT,
    const float* __restrict__ sinT, __bf16* __restrict__ Qg,
    __bf16* __restrict__ Kg, __bf16* __restrict__ Vt) {
  const int row = blockIdx.x;                 // b*S + s
  const int b = row >> 11, s = row & (NB_S - 1);
  for (int idx = threadIdx.x; idx < NB_NH * NB_DH; idx += blockDim.x) {
    const int h = idx >> 7, j = idx & 127;
    const size_t hrow = (size_t)(b * NB_NH + h) * NB_S + s;
    float qv, kv;
    if (j < NB_HALF) {
      qv = qc[(size_t)row * (NB_NH * NB_HALF) + h * NB_HALF + j];
      kv = kc[(size_t)row * (NB_NH * NB_HALF) + h * NB_HALF + j];
    } else {
      const int jj = j - NB_HALF;
      const int i = jj >> 1;
      const float c = cosT[(size_t)s * (NB_HALF / 2) + i];
      const float sn = sinT[(size_t)s * (NB_HALF / 2) + i];
      const float qx0 = qr[(size_t)row * (NB_NH * NB_HALF) + h * NB_HALF + 2 * i];
      const float qx1 = qr[(size_t)row * (NB_NH * NB_HALF) + h * NB_HALF + 2 * i + 1];
      qv = (jj & 1) ? (qx0 * sn + qx1 * c) : (qx0 * c - qx1 * sn);
      const float kx0 = kr[(size_t)row * NB_HALF + 2 * i];
      const float kx1 = kr[(size_t)row * NB_HALF + 2 * i + 1];
      kv = (jj & 1) ? (kx0 * sn + kx1 * c) : (kx0 * c - kx1 * sn);
    }
    Qg[hrow * NB_DH + j] = (__bf16)qv;
    Kg[hrow * NB_DH + j] = (__bf16)kv;
    Vt[((size_t)(b * NB_NH + h) * NB_DH + j) * NB_S + s] =
        (__bf16)vraw[(size_t)row * (NB_NH * NB_DH) + h * NB_DH + j];
  }
}

// ---------------------------------------------------------------------------
// Causal flash attention. Block = (b, h, 128 q-rows) = 8 waves x 16 rows.
// Per wave: Q in 4 v16bf frags; scores via WMMA (K frags contiguous from
// [.,S,128]); online softmax with shfl reductions inside 16-lane halves;
// P staged through wave-private LDS to reshape into A-fragments; PV via WMMA
// against transposed V. Output bf16 [B,S,NH*DH].
// ---------------------------------------------------------------------------
__global__ __launch_bounds__(256) void k_flash(const __bf16* __restrict__ Qg,
                                               const __bf16* __restrict__ Kg,
                                               const __bf16* __restrict__ Vt,
                                               __bf16* __restrict__ attn) {
  __shared__ __bf16 pL[8][16][64];
  const int lane = threadIdx.x & 31;
  const int wave = threadIdx.x >> 5;
  const int b = blockIdx.z, h = blockIdx.y;
  const int qbase = blockIdx.x * 128 + wave * 16;
  const int ls = lane & 15;
  const bool hiHalf = lane >= 16;
  const int half8 = hiHalf ? 8 : 0;
  const size_t headRow = (size_t)(b * NB_NH + h) * NB_S;
  const float scale = 0.08838834764831845f;  // 1/sqrt(128)

  v16bf qf[4];
  {
    const __bf16* qrow = Qg + (headRow + qbase + ls) * NB_DH;
#pragma unroll
    for (int c = 0; c < 4; ++c) {
      const int ak = c * 32 + half8;
      qf[c] = load_frag16(qrow + ak, qrow + ak + 16);
    }
  }

  float mrun[8], lrun[8];
  v8f of[8] = {};
#pragma unroll
  for (int r = 0; r < 8; ++r) { mrun[r] = -3.4e38f; lrun[r] = 0.f; }

  const int kmax = qbase + 16;  // causal bound (exclusive) for this wave
  for (int kb = 0; kb < kmax; kb += 64) {
    // ---- scores: 4 key tiles of 16 ----
    v8f sc[4] = {};
#pragma unroll
    for (int kt = 0; kt < 4; ++kt) {
      const __bf16* krow = Kg + (headRow + kb + kt * 16 + ls) * NB_DH;
#pragma unroll
      for (int c = 0; c < 4; ++c) {
        const int bk = c * 32 + half8;
        const v16bf kfrag = load_frag16(krow + bk, krow + bk + 16);
        sc[kt] = __builtin_amdgcn_wmma_f32_16x16x32_bf16(
            false, qf[c], false, kfrag, (short)0, sc[kt], false, false);
      }
    }
    // ---- scale + causal mask + running max ----
    float mnew[8];
#pragma unroll
    for (int r = 0; r < 8; ++r) mnew[r] = mrun[r];
#pragma unroll
    for (int kt = 0; kt < 4; ++kt) {
      const int col = kb + kt * 16 + ls;
#pragma unroll
      for (int r = 0; r < 8; ++r) {
        const int rowq = qbase + r + half8;
        const float v = (col <= rowq) ? sc[kt][r] * scale : -3.4e38f;
        sc[kt][r] = v;
        mnew[r] = fmaxf(mnew[r], v);
      }
    }
#pragma unroll
    for (int off = 1; off < 16; off <<= 1)
#pragma unroll
      for (int r = 0; r < 8; ++r)
        mnew[r] = fmaxf(mnew[r], __shfl_xor(mnew[r], off, 32));

    // ---- exp + row sums + rescale running state ----
    float alpha[8], psum[8];
#pragma unroll
    for (int r = 0; r < 8; ++r) { alpha[r] = __expf(mrun[r] - mnew[r]); psum[r] = 0.f; }
#pragma unroll
    for (int kt = 0; kt < 4; ++kt)
#pragma unroll
      for (int r = 0; r < 8; ++r) {
        const float pv = __expf(sc[kt][r] - mnew[r]);
        sc[kt][r] = pv;
        psum[r] += pv;
      }
#pragma unroll
    for (int off = 1; off < 16; off <<= 1)
#pragma unroll
      for (int r = 0; r < 8; ++r) psum[r] += __shfl_xor(psum[r], off, 32);
#pragma unroll
    for (int r = 0; r < 8; ++r) {
      lrun[r] = lrun[r] * alpha[r] + psum[r];
      mrun[r] = mnew[r];
    }
#pragma unroll
    for (int n = 0; n < 8; ++n)
#pragma unroll
      for (int r = 0; r < 8; ++r) of[n][r] = of[n][r] * alpha[r];

    // ---- P -> LDS (C layout) -> reload as A fragments ----
#pragma unroll
    for (int kt = 0; kt < 4; ++kt)
#pragma unroll
      for (int r = 0; r < 8; ++r)
        pL[wave][r + half8][kt * 16 + ls] = (__bf16)sc[kt][r];
    asm volatile("s_wait_dscnt 0" ::: "memory");  // wave-private LDS WAR/RAW

#pragma unroll
    for (int c2 = 0; c2 < 2; ++c2) {
      const __bf16* prow = &pL[wave][ls][c2 * 32 + half8];
      const v16bf pf = load_frag16(prow, prow + 16);
#pragma unroll
      for (int n = 0; n < 8; ++n) {
        const __bf16* vrow = Vt +
            ((size_t)(b * NB_NH + h) * NB_DH + n * 16 + ls) * NB_S +
            kb + c2 * 32 + half8;
        const v16bf vf = load_frag16(vrow, vrow + 16);
        of[n] = __builtin_amdgcn_wmma_f32_16x16x32_bf16(
            false, pf, false, vf, (short)0, of[n], false, false);
      }
    }
  }

  // ---- normalize + store bf16 [B, S, NH*DH] ----
#pragma unroll
  for (int r = 0; r < 8; ++r) lrun[r] = (lrun[r] > 0.f) ? 1.f / lrun[r] : 0.f;
#pragma unroll
  for (int n = 0; n < 8; ++n) {
    const int dh = n * 16 + ls;
#pragma unroll
    for (int r = 0; r < 8; ++r) {
      const int rowq = qbase + r + half8;
      attn[((size_t)b * NB_S + rowq) * (NB_NH * NB_DH) + h * NB_DH + dh] =
          (__bf16)(of[n][r] * lrun[r]);
    }
  }
}

// ---------------------------------------------------------------------------
// Host-side orchestration.
// ---------------------------------------------------------------------------
extern "C" void kernel_launch(void* const* d_in, const int* in_sizes, int n_in,
                              void* d_out, int out_size, void* d_ws, size_t ws_size,
                              hipStream_t stream) {
  const float* x     = (const float*)d_in[0];
  // d_in[1] = mask (static causal upper triangle) — handled analytically
  const float* cosT  = (const float*)d_in[2];
  const float* sinT  = (const float*)d_in[3];
  const float* w_in  = (const float*)d_in[4];
  const float* w_cq  = (const float*)d_in[5];
  const float* w_ckv = (const float*)d_in[6];
  const float* W_cq  = (const float*)d_in[7];
  const float* W_q   = (const float*)d_in[8];
  const float* W_qR  = (const float*)d_in[9];
  const float* W_ckv = (const float*)d_in[10];
  const float* W_k   = (const float*)d_in[11];
  const float* W_kR  = (const float*)d_in[12];
  const float* W_v   = (const float*)d_in[13];
  const float* W_o   = (const float*)d_in[14];

  char* ws = (char*)d_ws;
  size_t off = 0;
  auto carve = [&](size_t bytes) -> void* {
    void* p = ws + off;
    off += (bytes + 255) & ~(size_t)255;
    return p;
  };

  // bf16 weight transposes (N x K)
  __bf16* cq_t  = (__bf16*)carve((size_t)NB_DCQ * NB_D * 2);
  __bf16* q_t   = (__bf16*)carve((size_t)(NB_NH * NB_HALF) * NB_DCQ * 2);
  __bf16* qR_t  = (__bf16*)carve((size_t)(NB_NH * NB_HALF) * NB_DCQ * 2);
  __bf16* ckv_t = (__bf16*)carve((size_t)NB_DCKV * NB_D * 2);
  __bf16* k_t   = (__bf16*)carve((size_t)(NB_NH * NB_HALF) * NB_DCKV * 2);
  __bf16* kR_t  = (__bf16*)carve((size_t)NB_HALF * NB_D * 2);
  __bf16* v_t   = (__bf16*)carve((size_t)(NB_NH * NB_DH) * NB_DCKV * 2);
  __bf16* o_t   = (__bf16*)carve((size_t)NB_D * (NB_NH * NB_DH) * 2);
  // bf16 activations
  __bf16* hbf    = (__bf16*)carve((size_t)NB_M * NB_D * 2);
  __bf16* cq_bf  = (__bf16*)carve((size_t)NB_M * NB_DCQ * 2);
  __bf16* ckv_bf = (__bf16*)carve((size_t)NB_M * NB_DCKV * 2);
  __bf16* Qg     = (__bf16*)carve((size_t)NB_M * NB_NH * NB_DH * 2);  // [B,NH,S,128]
  __bf16* Kg     = (__bf16*)carve((size_t)NB_M * NB_NH * NB_DH * 2);
  __bf16* Vt     = (__bf16*)carve((size_t)NB_M * NB_NH * NB_DH * 2);
  __bf16* attnb  = (__bf16*)carve((size_t)NB_M * NB_NH * NB_DH * 2);
  // fp32 GEMM outputs
  float* cq_raw  = (float*)carve((size_t)NB_M * NB_DCQ * 4);
  float* ckv_raw = (float*)carve((size_t)NB_M * NB_DCKV * 4);
  float* qc_raw  = (float*)carve((size_t)NB_M * NB_NH * NB_HALF * 4);
  float* qr_raw  = (float*)carve((size_t)NB_M * NB_NH * NB_HALF * 4);
  float* kc_raw  = (float*)carve((size_t)NB_M * NB_NH * NB_HALF * 4);
  float* kr_raw  = (float*)carve((size_t)NB_M * NB_HALF * 4);
  float* v_raw   = (float*)carve((size_t)NB_M * NB_NH * NB_DH * 4);
  (void)ws_size; (void)in_sizes; (void)n_in; (void)out_size;

  auto launchT = [&](const float* s, __bf16* d, int K, int N) {
    const int total = K * N;
    k_transpose_cast<<<(total + 255) / 256, 256, 0, stream>>>(s, d, K, N);
  };
  launchT(W_cq, cq_t, NB_D, NB_DCQ);
  launchT(W_q,  q_t,  NB_DCQ, NB_NH * NB_HALF);
  launchT(W_qR, qR_t, NB_DCQ, NB_NH * NB_HALF);
  launchT(W_ckv, ckv_t, NB_D, NB_DCKV);
  launchT(W_k,  k_t,  NB_DCKV, NB_NH * NB_HALF);
  launchT(W_kR, kR_t, NB_D, NB_HALF);
  launchT(W_v,  v_t,  NB_DCKV, NB_NH * NB_DH);
  launchT(W_o,  o_t,  NB_NH * NB_DH, NB_D);

  // h = rms(x) in bf16
  k_rms_bf16<<<NB_M, 256, 0, stream>>>(x, w_in, hbf, NB_D);

  auto gemm = [&](const __bf16* A, const __bf16* Bt, float* C,
                  const float* resid, int M_, int N_, int K_) {
    dim3 grid(N_ / 64, M_ / 256);
    k_gemm_bf16<<<grid, 256, 0, stream>>>(A, Bt, C, resid, M_, N_, K_);
  };

  gemm(hbf, cq_t,  cq_raw,  nullptr, NB_M, NB_DCQ, NB_D);
  gemm(hbf, ckv_t, ckv_raw, nullptr, NB_M, NB_DCKV, NB_D);
  gemm(hbf, kR_t,  kr_raw,  nullptr, NB_M, NB_HALF, NB_D);

  k_rms_bf16<<<NB_M, 256, 0, stream>>>(cq_raw,  w_cq,  cq_bf,  NB_DCQ);
  k_rms_bf16<<<NB_M, 256, 0, stream>>>(ckv_raw, w_ckv, ckv_bf, NB_DCKV);

  gemm(cq_bf,  q_t,  qc_raw, nullptr, NB_M, NB_NH * NB_HALF, NB_DCQ);
  gemm(cq_bf,  qR_t, qr_raw, nullptr, NB_M, NB_NH * NB_HALF, NB_DCQ);
  gemm(ckv_bf, k_t,  kc_raw, nullptr, NB_M, NB_NH * NB_HALF, NB_DCKV);
  gemm(ckv_bf, v_t,  v_raw,  nullptr, NB_M, NB_NH * NB_DH,   NB_DCKV);

  k_build_qkv<<<NB_M, 256, 0, stream>>>(qc_raw, qr_raw, kc_raw, kr_raw, v_raw,
                                        cosT, sinT, Qg, Kg, Vt);

  dim3 fgrid(NB_S / 128, NB_NH, NB_B);
  k_flash<<<fgrid, 256, 0, stream>>>(Qg, Kg, Vt, attnb);

  // out = x + attn @ W_o  (fp32 output)
  gemm(attnb, o_t, (float*)d_out, x, NB_M, NB_D, NB_NH * NB_DH);
}
